// NoBrainEncoderBlock_17781164606013
// MI455X (gfx1250) — compile-verified, matrix operand order checked by reading, not run on hardware
//
#include <hip/hip_runtime.h>
#include <hip/hip_bf16.h>
#include <math.h>

#define B_DIM 8192
#define N_DIM 8192
#define D_DIM 768

typedef _Float16 v16h __attribute__((ext_vector_type(16)));
typedef _Float16 v8h  __attribute__((ext_vector_type(8)));
typedef float    v8f  __attribute__((ext_vector_type(8)));

// ---------------------------------------------------------------------------
// Stage 0: zero the row-sum accumulators (must happen every call: the harness
// replays without re-poisoning, and we use atomicAdd accumulation).
// ---------------------------------------------------------------------------
__global__ void zero_f32_kernel(float* __restrict__ p, int n) {
    int i = blockIdx.x * blockDim.x + threadIdx.x;
    if (i < n) p[i] = 0.0f;
}

// ---------------------------------------------------------------------------
// Stage 1: L2-normalize each 768-elem row, fp32 -> f16. One 256-thread block
// per row; 3 elements per thread; LDS tree reduction for sum of squares.
// ---------------------------------------------------------------------------
__global__ void normalize_rows_kernel(const float* __restrict__ src,
                                      _Float16* __restrict__ dst) {
    __shared__ float red[256];
    const int row = blockIdx.x;
    const int tid = threadIdx.x;
    const float* r = src + (size_t)row * D_DIM;
    float v0 = r[tid];
    float v1 = r[tid + 256];
    float v2 = r[tid + 512];
    red[tid] = v0 * v0 + v1 * v1 + v2 * v2;
    __syncthreads();
    #pragma unroll
    for (int off = 128; off > 0; off >>= 1) {
        if (tid < off) red[tid] += red[tid + off];
        __syncthreads();
    }
    const float nrm   = sqrtf(red[0]);
    const float scale = 1.0f / fmaxf(nrm, 1e-12f);
    _Float16* o = dst + (size_t)row * D_DIM;
    o[tid]       = (_Float16)(v0 * scale);
    o[tid + 256] = (_Float16)(v1 * scale);
    o[tid + 512] = (_Float16)(v2 * scale);
}

// ---------------------------------------------------------------------------
// Fragment load matching the CDNA5 16-bit WMMA A/B VGPR layout:
//   lane <16 : row/col = base+lane,    halves = K{0..7}   then K{16..23}
//   lane>=16 : row/col = base+lane-16, halves = K{8..15}  then K{24..31}
// Two 16B global loads per fragment. A (q, M x K) and B (k-as-columns, K x N)
// use the identical pattern because both q and k are row-major over D.
// ---------------------------------------------------------------------------
__device__ __forceinline__ v16h load_frag(const _Float16* __restrict__ p,
                                          int tbase, int kk, int lane) {
    const int r = tbase + (lane & 15);
    const _Float16* base = p + (size_t)r * D_DIM + kk + ((lane & 16) ? 8 : 0);
    v8h lo = *(const v8h*)(base);
    v8h hi = *(const v8h*)(base + 16);
    return __builtin_shufflevector(lo, hi,
        0, 1, 2, 3, 4, 5, 6, 7, 8, 9, 10, 11, 12, 13, 14, 15);
}

// ---------------------------------------------------------------------------
// Stage 2: dual GEMM (s1 = q1n.k1n^T, s2 = q2n.k2n^T) with fused
// mask -> clip[0,1] -> exp epilogue and atomic per-row partial sums.
// Scores are clipped to [0,1], so softmax needs no max-subtraction: exp is
// always safe. Block tile = 128(M) x 64(N); 8 waves in a 4x2 grid; each wave
// computes a 32x32 tile for BOTH gemms (2x2 accum tiles x 2 = 64 accum VGPRs).
// ---------------------------------------------------------------------------
__global__ void __launch_bounds__(256)
gemm_dual_exp_kernel(const _Float16* __restrict__ q1,
                     const _Float16* __restrict__ k1,
                     const _Float16* __restrict__ q2,
                     const _Float16* __restrict__ k2,
                     const float* __restrict__ mask,
                     float* __restrict__ e1,
                     float* __restrict__ e2,
                     float* __restrict__ sum1,
                     float* __restrict__ sum2) {
    const int lane = threadIdx.x & 31;
    const int wave = threadIdx.x >> 5;            // 0..7
    const int wr   = wave & 3;                    // 4 row-waves
    const int wc   = wave >> 2;                   // 2 col-waves
    const int rowBase = blockIdx.y * 128 + wr * 32;
    const int colBase = blockIdx.x * 64  + wc * 32;

    v8f c1[2][2] = {};
    v8f c2[2][2] = {};

    for (int kk = 0; kk < D_DIM; kk += 32) {
        v16h a1[2], a2[2], b1[2], b2[2];
        #pragma unroll
        for (int i = 0; i < 2; ++i) {
            a1[i] = load_frag(q1, rowBase + i * 16, kk, lane);
            a2[i] = load_frag(q2, rowBase + i * 16, kk, lane);
        }
        #pragma unroll
        for (int j = 0; j < 2; ++j) {
            b1[j] = load_frag(k1, colBase + j * 16, kk, lane);
            b2[j] = load_frag(k2, colBase + j * 16, kk, lane);
        }
        #pragma unroll
        for (int i = 0; i < 2; ++i) {
            #pragma unroll
            for (int j = 0; j < 2; ++j) {
                c1[i][j] = __builtin_amdgcn_wmma_f32_16x16x32_f16(
                    false, a1[i], false, b1[j], (short)0, c1[i][j], false, false);
                c2[i][j] = __builtin_amdgcn_wmma_f32_16x16x32_f16(
                    false, a2[i], false, b2[j], (short)0, c2[i][j], false, false);
            }
        }
    }

    // Epilogue. C/D layout: VGPR v, lane L -> M = v + 8*(L>=16), N = L&15.
    const int laneLo = lane & 15;
    const int mOff   = (lane >> 4) * 8;
    #pragma unroll
    for (int i = 0; i < 2; ++i) {
        #pragma unroll
        for (int v = 0; v < 8; ++v) {
            const int    m      = rowBase + i * 16 + mOff + v;
            const size_t rowoff = (size_t)m * N_DIM;
            float acc1 = 0.0f, acc2 = 0.0f;
            #pragma unroll
            for (int j = 0; j < 2; ++j) {
                const int n  = colBase + j * 16 + laneLo;
                const float mk = mask[rowoff + n];
                float s1 = fminf(fmaxf(c1[i][j][v] * mk, 0.0f), 1.0f);
                float s2 = fminf(fmaxf(c2[i][j][v] * mk, 0.0f), 1.0f);
                float x1 = __expf(s1);
                float x2 = __expf(s2);
                e1[rowoff + n] = x1;
                e2[rowoff + n] = x2;
                acc1 += x1;
                acc2 += x2;
            }
            // Reduce across the 16 lanes sharing row m (xor<=8 stays in-group).
            #pragma unroll
            for (int off = 8; off > 0; off >>= 1) {
                acc1 += __shfl_xor(acc1, off, 32);
                acc2 += __shfl_xor(acc2, off, 32);
            }
            if (laneLo == 0) {
                atomicAdd(&sum1[m], acc1);
                atomicAdd(&sum2[m], acc2);
            }
        }
    }
}

// ---------------------------------------------------------------------------
// Stage 3: out = a*e1/sum1[row] + (1-a)*e2/sum2[row], a = sigmoid(temp).
// float4 vectorized; e1 aliases out (same element read-then-write per thread).
// ---------------------------------------------------------------------------
__global__ void finalize_kernel(const float* e1,
                                const float* __restrict__ e2,
                                const float* __restrict__ sum1,
                                const float* __restrict__ sum2,
                                const float* __restrict__ temp,
                                float* out) {
    const int m  = blockIdx.y;
    const int n0 = (blockIdx.x * blockDim.x + threadIdx.x) * 4;
    const float t = temp[0];
    const float a = 1.0f / (1.0f + __expf(-t));
    const float i1 = a / sum1[m];
    const float i2 = (1.0f - a) / sum2[m];
    const size_t off = (size_t)m * N_DIM + n0;
    float4 x1 = *(const float4*)(e1 + off);
    float4 x2 = *(const float4*)(e2 + off);
    float4 r;
    r.x = x1.x * i1 + x2.x * i2;
    r.y = x1.y * i1 + x2.y * i2;
    r.z = x1.z * i1 + x2.z * i2;
    r.w = x1.w * i1 + x2.w * i2;
    *(float4*)(out + off) = r;
}

// ---------------------------------------------------------------------------
extern "C" void kernel_launch(void* const* d_in, const int* in_sizes, int n_in,
                              void* d_out, int out_size, void* d_ws, size_t ws_size,
                              hipStream_t stream) {
    const float* q1   = (const float*)d_in[0];
    const float* k1   = (const float*)d_in[1];
    const float* q2   = (const float*)d_in[2];
    const float* k2   = (const float*)d_in[3];
    const float* mask = (const float*)d_in[4];
    const float* temp = (const float*)d_in[5];
    float* out = (float*)d_out;

    // Workspace layout (256B aligned slices).
    char*  ws  = (char*)d_ws;
    size_t off = 0;
    auto alloc = [&](size_t bytes) -> char* {
        char* p = ws + off;
        off += (bytes + 255) & ~(size_t)255;
        return p;
    };
    const size_t halfMat = (size_t)B_DIM * D_DIM * sizeof(_Float16);
    _Float16* q1h  = (_Float16*)alloc(halfMat);
    _Float16* k1h  = (_Float16*)alloc(halfMat);
    _Float16* q2h  = (_Float16*)alloc(halfMat);
    _Float16* k2h  = (_Float16*)alloc(halfMat);
    float*    sums = (float*)alloc((size_t)2 * B_DIM * sizeof(float));
    float*    sum1 = sums;
    float*    sum2 = sums + B_DIM;
    float*    e2   = (float*)alloc((size_t)B_DIM * N_DIM * sizeof(float));
    float*    e1   = out;  // reuse d_out as the e1 scratch; finalize overwrites.

    // Stage 0: zero accumulators (every call: deterministic replay).
    zero_f32_kernel<<<(2 * B_DIM + 255) / 256, 256, 0, stream>>>(sums, 2 * B_DIM);

    // Stage 1: normalize fp32 -> f16.
    normalize_rows_kernel<<<B_DIM, 256, 0, stream>>>(q1, q1h);
    normalize_rows_kernel<<<N_DIM, 256, 0, stream>>>(k1, k1h);
    normalize_rows_kernel<<<B_DIM, 256, 0, stream>>>(q2, q2h);
    normalize_rows_kernel<<<N_DIM, 256, 0, stream>>>(k2, k2h);

    // Stage 2: dual WMMA GEMM + fused mask/clip/exp + row-sum atomics.
    dim3 ggrid(N_DIM / 64, B_DIM / 128);
    gemm_dual_exp_kernel<<<ggrid, 256, 0, stream>>>(
        q1h, k1h, q2h, k2h, mask, e1, e2, sum1, sum2);

    // Stage 3: normalize + blend.
    dim3 fgrid(N_DIM / (256 * 4), B_DIM);
    finalize_kernel<<<fgrid, 256, 0, stream>>>(e1, e2, sum1, sum2, temp, out);
}
